// DecoderLayer_82265803588426
// MI455X (gfx1250) — matrix operand order repaired and checked by reading, not compiled, not run on hardware
//
#include <hip/hip_runtime.h>
#include <hip/hip_bf16.h>
#include <math.h>

// ---- shapes (fixed by the reference) ----
#define D_MODEL  1024
#define N_HEADS  16
#define HEAD_DIM 64
#define FFN_H    4096
#define BATCH    4
#define SEQ      1024
#define ROWS     (BATCH * SEQ)   // 4096

typedef __attribute__((ext_vector_type(16))) __bf16 v16bf;
typedef __attribute__((ext_vector_type(8)))  __bf16 v8bf;
typedef __attribute__((ext_vector_type(2)))  __bf16 v2bf;
typedef __attribute__((ext_vector_type(8)))  float  v8f;
typedef __attribute__((ext_vector_type(4)))  float  v4f;

__device__ __forceinline__ __bf16 to_bf16(float f) { return (__bf16)f; }

__device__ __forceinline__ v16bf cat16(v8bf lo, v8bf hi) {
    return __builtin_shufflevector(lo, hi, 0,1,2,3,4,5,6,7,8,9,10,11,12,13,14,15);
}

// =====================================================================
// bf16 WMMA GEMM:  C[M,N] = A[M,K] * B[K,N] (+bias) (ReLU?)
// REQUIRES: M%64==0, N%64==0, K%32==0 (true for all call sites).
// 128 threads = 4 waves; block tile 64x64; wave tile 32x32; K-step 32.
// A staged row-major in LDS; B staged TRANSPOSED in LDS (k-pairs packed
// into b32 stores) so each lane's B fragment is two ds_load_b128.
// =====================================================================
__global__ __launch_bounds__(128)
void gemm_bf16_wmma(const __bf16* __restrict__ A, const __bf16* __restrict__ B,
                    const float* __restrict__ bias,
                    float* __restrict__ outF, __bf16* __restrict__ outBF,
                    int M, int N, int K, int relu)
{
    __shared__ __bf16 As[64 * 32];   // [row][k]   4 KB
    __shared__ __bf16 Bt[64 * 32];   // [col][k]   4 KB (transposed tile)

    const int tid  = threadIdx.x;
    const int lane = tid & 31;
    const int wave = tid >> 5;
    const int wm   = wave >> 1;
    const int wn   = wave & 1;
    const int bm   = blockIdx.y * 64;
    const int bn   = blockIdx.x * 64;
    const int half = lane >> 4;
    const int l15  = lane & 15;

    // staging coordinates
    const int ar = tid >> 1;            // A: row 0..63
    const int ac = (tid & 1) * 16;      // A: k base {0,16}
    const int bk = (tid & 15) * 2;      // B: k row pair base 0,2,..,30
    const int bc = (tid >> 4) * 8;      // B: col base 0,8,..,56

    v8f acc[2][2];
    const v8f vzero = {0.f,0.f,0.f,0.f,0.f,0.f,0.f,0.f};
#pragma unroll
    for (int i = 0; i < 2; ++i)
#pragma unroll
        for (int j = 0; j < 2; ++j) acc[i][j] = vzero;

    for (int k0 = 0; k0 < K; k0 += 32) {
        // ---- stage A tile 64x32 : 128-bit loads, straight copy ----
        {
            const __bf16* ga = A + (long)(bm + ar) * K + k0 + ac;
            v8bf v0 = *(const v8bf*)(ga);
            v8bf v1 = *(const v8bf*)(ga + 8);
            *(v8bf*)(&As[ar * 32 + ac])     = v0;
            *(v8bf*)(&As[ar * 32 + ac + 8]) = v1;
        }
        // ---- stage B tile 32x64 transposed: 2 k-rows x 8 cols per thread,
        //      k-pairs packed into b32 LDS stores ----
        {
            const __bf16* gb = B + (long)(k0 + bk) * N + bn + bc;
            v8bf r0 = *(const v8bf*)(gb);
            v8bf r1 = *(const v8bf*)(gb + N);
#pragma unroll
            for (int j = 0; j < 8; ++j) {
                v2bf pk = {r0[j], r1[j]};
                *(v2bf*)(&Bt[(bc + j) * 32 + bk]) = pk;
            }
        }
        // prefetch next K tile while we compute on this one
        if (k0 + 32 < K) {
            __builtin_prefetch(A + (long)(bm + ar) * K + k0 + 32 + ac, 0, 1);
            __builtin_prefetch(B + (long)(k0 + 32 + bk) * N + bn + bc, 0, 1);
        }
        __syncthreads();

        // ---- fragments ----
        v16bf afrag[2], bfrag[2];
#pragma unroll
        for (int ms = 0; ms < 2; ++ms) {
            const v8bf* ap = (const v8bf*)&As[(wm * 32 + ms * 16 + l15) * 32];
            afrag[ms] = cat16(ap[half], ap[2 + half]);      // K: 8h..+7 | 16+8h..+7
        }
#pragma unroll
        for (int ns = 0; ns < 2; ++ns) {
            const v8bf* bp = (const v8bf*)&Bt[(wn * 32 + ns * 16 + l15) * 32];
            bfrag[ns] = cat16(bp[half], bp[2 + half]); // k: 8h..+7 | 16+8h..+7
        }
#pragma unroll
        for (int ms = 0; ms < 2; ++ms)
#pragma unroll
            for (int ns = 0; ns < 2; ++ns)
                acc[ms][ns] = __builtin_amdgcn_wmma_f32_16x16x32_bf16(
                    false, afrag[ms], false, bfrag[ns],
                    (short)0, acc[ms][ns], false, false);
        __syncthreads();
    }

    // ---- epilogue ----
#pragma unroll
    for (int ms = 0; ms < 2; ++ms)
#pragma unroll
        for (int ns = 0; ns < 2; ++ns) {
            const int colg = bn + wn * 32 + ns * 16 + l15;
            const float bv = bias ? bias[colg] : 0.f;
#pragma unroll
            for (int r = 0; r < 8; ++r) {
                const int rowg = bm + wm * 32 + ms * 16 + half * 8 + r;
                float v = acc[ms][ns][r] + bv;
                if (relu) v = v > 0.f ? v : 0.f;
                if (outF)  outF[(long)rowg * N + colg] = v;
                if (outBF) outBF[(long)rowg * N + colg] = to_bf16(v);
            }
        }
}

// =====================================================================
// Attention: one wave per (b*H+h, 16-query-row block).
// Q,K bf16 [B,H,S,hd]; V bf16 TRANSPOSED [B,H,hd,S] so the P@V B-fragment
// is two contiguous global_load_b128.  Scores strip 16xS fp32 in LDS,
// softmax in LDS, then P@V via WMMA with 4 accumulators.
// =====================================================================
__global__ __launch_bounds__(32)
void attention_wmma(const __bf16* __restrict__ Q, const __bf16* __restrict__ Km,
                    const __bf16* __restrict__ Vt, const float* __restrict__ mask,
                    float* __restrict__ Ovals, float scale)
{
    __shared__ float sS[16 * SEQ];            // 64 KB
    const int lane = threadIdx.x & 31;
    const int half = lane >> 4;
    const int l15  = lane & 15;
    const int bh   = blockIdx.y;              // b*H + h
    const int q0   = blockIdx.x * 16;         // query row base
    const long base = (long)bh * SEQ * HEAD_DIM;
    const v8f vzero = {0.f,0.f,0.f,0.f,0.f,0.f,0.f,0.f};

    // ---- scores = Q @ K^T ----
    for (int kt = 0; kt < SEQ / 16; ++kt) {
        v8f acc = vzero;
#pragma unroll
        for (int kk = 0; kk < HEAD_DIM; kk += 32) {
            // A fragment: Q row, two contiguous 8-elem runs
            const v8bf* qp = (const v8bf*)(Q + base + (long)(q0 + l15) * HEAD_DIM + kk);
            v16bf a = cat16(qp[half], qp[2 + half]);
            // B fragment: K^T -> one K row, two 8-elem runs (k 8h..+7 | 16+8h..+7)
            const v8bf* kp = (const v8bf*)(Km + base + (long)(kt * 16 + l15) * HEAD_DIM + kk);
            v16bf b = cat16(kp[half], kp[2 + half]);
            acc = __builtin_amdgcn_wmma_f32_16x16x32_bf16(
                false, a, false, b, (short)0, acc, false, false);
        }
#pragma unroll
        for (int r = 0; r < 8; ++r) {
            int rl = half * 8 + r;
            int cg = kt * 16 + l15;
            float s = acc[r] * scale;
            if (mask) s += mask[(long)(q0 + rl) * SEQ + cg];
            sS[rl * SEQ + cg] = s;
        }
    }
    __syncthreads();

    // ---- row softmax (lanes 0..15, one row each) ----
    if (lane < 16) {
        float* rowp = &sS[lane * SEQ];
        float m = rowp[0];
        for (int j = 1; j < SEQ; ++j) m = fmaxf(m, rowp[j]);
        float s = 0.f;
        for (int j = 0; j < SEQ; ++j) { float e = __expf(rowp[j] - m); rowp[j] = e; s += e; }
        float inv = 1.f / s;
        for (int j = 0; j < SEQ; ++j) rowp[j] *= inv;
    }
    __syncthreads();

    // ---- out = P @ V : P fragment built once per k-step, 4 accumulators ----
    v8f oacc[4];
#pragma unroll
    for (int nt = 0; nt < 4; ++nt) oacc[nt] = vzero;

    for (int kt = 0; kt < SEQ; kt += 32) {
        // A fragment from fp32 probabilities in LDS (vector loads + convert)
        const v4f* sp = (const v4f*)&sS[l15 * SEQ + kt];
        v4f f0 = sp[2 * half];
        v4f f1 = sp[2 * half + 1];
        v4f f2 = sp[4 + 2 * half];
        v4f f3 = sp[4 + 2 * half + 1];
        v16bf a;
#pragma unroll
        for (int i = 0; i < 4; ++i) {
            a[i]      = to_bf16(f0[i]);
            a[4 + i]  = to_bf16(f1[i]);
            a[8 + i]  = to_bf16(f2[i]);
            a[12 + i] = to_bf16(f3[i]);
        }
#pragma unroll
        for (int nt = 0; nt < 4; ++nt) {
            // B fragment: V^T row (dim nt*16+l15), contiguous seq -> vector loads
            const v8bf* vp = (const v8bf*)(Vt + base + (long)(nt * 16 + l15) * SEQ + kt);
            v16bf b = cat16(vp[half], vp[2 + half]);
            oacc[nt] = __builtin_amdgcn_wmma_f32_16x16x32_bf16(
                false, a, false, b, (short)0, oacc[nt], false, false);
        }
    }
#pragma unroll
    for (int nt = 0; nt < 4; ++nt)
#pragma unroll
        for (int r = 0; r < 8; ++r) {
            int rl = half * 8 + r;
            Ovals[base + (long)(q0 + rl) * HEAD_DIM + nt * 16 + l15] = oacc[nt][r];
        }
}

// =====================================================================
// Elementwise / reshape helpers
// =====================================================================
__global__ void f32_to_bf16(const float* __restrict__ in, __bf16* __restrict__ out, long n)
{
    long i = (long)blockIdx.x * blockDim.x + threadIdx.x;
    long stride = (long)gridDim.x * blockDim.x;
    for (; i < n; i += stride) out[i] = to_bf16(in[i]);
}

// qkv fp32 [B*S, 3D] -> q,k bf16 [B,H,S,hd]; v bf16 transposed [B,H,hd,S]
__global__ void split_qkv(const float* __restrict__ qkv,
                          __bf16* __restrict__ q, __bf16* __restrict__ k, __bf16* __restrict__ vt)
{
    long i = (long)blockIdx.x * blockDim.x + threadIdx.x;
    const long total = (long)BATCH * N_HEADS * SEQ * HEAD_DIM;
    if (i >= total) return;
    int d = (int)(i % HEAD_DIM); long t = i / HEAD_DIM;
    int s = (int)(t % SEQ); t /= SEQ;
    int h = (int)(t % N_HEADS); int b = (int)(t / N_HEADS);
    long rowbase = ((long)b * SEQ + s) * (3 * D_MODEL) + (long)h * 3 * HEAD_DIM;
    q[i] = to_bf16(qkv[rowbase + d]);
    k[i] = to_bf16(qkv[rowbase + HEAD_DIM + d]);
    vt[(((long)b * N_HEADS + h) * HEAD_DIM + d) * SEQ + s]
        = to_bf16(qkv[rowbase + 2 * HEAD_DIM + d]);
}

// kv fp32 [B*S, 2D] -> k bf16 [B,H,S,hd]; v bf16 transposed [B,H,hd,S]
__global__ void split_kv(const float* __restrict__ kv,
                         __bf16* __restrict__ k, __bf16* __restrict__ vt)
{
    long i = (long)blockIdx.x * blockDim.x + threadIdx.x;
    const long total = (long)BATCH * N_HEADS * SEQ * HEAD_DIM;
    if (i >= total) return;
    int d = (int)(i % HEAD_DIM); long t = i / HEAD_DIM;
    int s = (int)(t % SEQ); t /= SEQ;
    int h = (int)(t % N_HEADS); int b = (int)(t / N_HEADS);
    long rowbase = ((long)b * SEQ + s) * (2 * D_MODEL) + (long)h * 2 * HEAD_DIM;
    k[i] = to_bf16(kv[rowbase + d]);
    vt[(((long)b * N_HEADS + h) * HEAD_DIM + d) * SEQ + s]
        = to_bf16(kv[rowbase + HEAD_DIM + d]);
}

// qproj fp32 [B*S, D] -> q bf16 [B,H,S,hd]
__global__ void split_q(const float* __restrict__ qp, __bf16* __restrict__ q)
{
    long i = (long)blockIdx.x * blockDim.x + threadIdx.x;
    const long total = (long)BATCH * N_HEADS * SEQ * HEAD_DIM;
    if (i >= total) return;
    int d = (int)(i % HEAD_DIM); long t = i / HEAD_DIM;
    int s = (int)(t % SEQ); t /= SEQ;
    int h = (int)(t % N_HEADS); int b = (int)(t / N_HEADS);
    q[i] = to_bf16(qp[((long)b * SEQ + s) * D_MODEL + (long)h * HEAD_DIM + d]);
}

// vals fp32 [B,H,S,hd] -> bf16 [B,S,H*hd] (proper transpose, cross-attn path)
__global__ void transpose_vals(const float* __restrict__ vals, __bf16* __restrict__ out)
{
    long i = (long)blockIdx.x * blockDim.x + threadIdx.x;
    const long total = (long)ROWS * D_MODEL;
    if (i >= total) return;
    int dd = (int)(i % D_MODEL);
    int s  = (int)((i / D_MODEL) % SEQ);
    int b  = (int)(i / ((long)SEQ * D_MODEL));
    int h  = dd / HEAD_DIM;
    int d  = dd % HEAD_DIM;
    out[i] = to_bf16(vals[(((long)b * N_HEADS + h) * SEQ + s) * HEAD_DIM + d]);
}

// LayerNorm over D=1024 with fused residual: t = z+res
__global__ __launch_bounds__(256)
void layernorm_kernel(const float* __restrict__ z, const float* __restrict__ res,
                      const float* __restrict__ gamma, const float* __restrict__ beta,
                      float* __restrict__ outF, __bf16* __restrict__ outBF)
{
    __shared__ float red[256];
    __shared__ float red2[256];
    const int row = blockIdx.x;
    const int tid = threadIdx.x;
    const float* zp = z   + (long)row * D_MODEL;
    const float* rp = res + (long)row * D_MODEL;
    float vals[4]; float s = 0.f, s2 = 0.f;
#pragma unroll
    for (int i = 0; i < 4; ++i) {
        float t = zp[tid + i * 256] + rp[tid + i * 256];
        vals[i] = t; s += t; s2 += t * t;
    }
    red[tid] = s; red2[tid] = s2; __syncthreads();
    for (int off = 128; off > 0; off >>= 1) {
        if (tid < off) { red[tid] += red[tid + off]; red2[tid] += red2[tid + off]; }
        __syncthreads();
    }
    float mean = red[0] * (1.f / D_MODEL);
    float var  = red2[0] * (1.f / D_MODEL) - mean * mean;
    float rstd = rsqrtf(var + 1e-5f);
#pragma unroll
    for (int i = 0; i < 4; ++i) {
        int c = tid + i * 256;
        float o = gamma[c] * (vals[i] - mean) * rstd + beta[c];
        if (outF)  outF[(long)row * D_MODEL + c] = o;
        if (outBF) outBF[(long)row * D_MODEL + c] = to_bf16(o);
    }
}

// =====================================================================
// Orchestration
// =====================================================================
extern "C" void kernel_launch(void* const* d_in, const int* in_sizes, int n_in,
                              void* d_out, int out_size, void* d_ws, size_t ws_size,
                              hipStream_t stream)
{
    const float* x     = (const float*)d_in[0];
    const float* y     = (const float*)d_in[1];
    const float* dmask = (const float*)d_in[2];
    const float* w_qkv = (const float*)d_in[3];
    const float* b_qkv = (const float*)d_in[4];
    const float* w_so  = (const float*)d_in[5];
    const float* b_so  = (const float*)d_in[6];
    const float* g1    = (const float*)d_in[7];
    const float* be1   = (const float*)d_in[8];
    const float* w_kv  = (const float*)d_in[9];
    const float* b_kv  = (const float*)d_in[10];
    const float* w_q   = (const float*)d_in[11];
    const float* b_q   = (const float*)d_in[12];
    const float* w_co  = (const float*)d_in[13];
    const float* b_co  = (const float*)d_in[14];
    const float* g2    = (const float*)d_in[15];
    const float* be2   = (const float*)d_in[16];
    const float* w_f1  = (const float*)d_in[17];
    const float* b_f1  = (const float*)d_in[18];
    const float* w_f2  = (const float*)d_in[19];
    const float* b_f2  = (const float*)d_in[20];
    const float* g3    = (const float*)d_in[21];
    const float* be3   = (const float*)d_in[22];
    float* out = (float*)d_out;

    // carve workspace
    char* p = (char*)d_ws;
    auto alloc = [&](size_t bytes) -> char* {
        char* r = p; p += (bytes + 255) & ~(size_t)255; return r;
    };
    __bf16* wqkv_bf = (__bf16*)alloc((size_t)D_MODEL * 3 * D_MODEL * 2);
    __bf16* wso_bf  = (__bf16*)alloc((size_t)D_MODEL * D_MODEL * 2);
    __bf16* wkv_bf  = (__bf16*)alloc((size_t)D_MODEL * 2 * D_MODEL * 2);
    __bf16* wq_bf   = (__bf16*)alloc((size_t)D_MODEL * D_MODEL * 2);
    __bf16* wco_bf  = (__bf16*)alloc((size_t)D_MODEL * D_MODEL * 2);
    __bf16* wf1_bf  = (__bf16*)alloc((size_t)D_MODEL * FFN_H * 2);
    __bf16* wf2_bf  = (__bf16*)alloc((size_t)FFN_H * D_MODEL * 2);
    __bf16* xbf     = (__bf16*)alloc((size_t)ROWS * D_MODEL * 2);
    __bf16* actbf   = (__bf16*)alloc((size_t)ROWS * D_MODEL * 2);
    float*  qkvf    = (float*)alloc((size_t)ROWS * 3 * D_MODEL * 4); // also reused for kv
    __bf16* qbf     = (__bf16*)alloc((size_t)ROWS * D_MODEL * 2);
    __bf16* kbf     = (__bf16*)alloc((size_t)ROWS * D_MODEL * 2);
    __bf16* vbf     = (__bf16*)alloc((size_t)ROWS * D_MODEL * 2);   // V^T [B,H,hd,S]
    float*  valsf   = (float*)alloc((size_t)ROWS * D_MODEL * 4);
    float*  zf      = (float*)alloc((size_t)ROWS * D_MODEL * 4);
    float*  y1f     = (float*)alloc((size_t)ROWS * D_MODEL * 4);
    __bf16* y1bf    = (__bf16*)alloc((size_t)ROWS * D_MODEL * 2);
    float*  y2f     = (float*)alloc((size_t)ROWS * D_MODEL * 4);
    __bf16* y2bf    = (__bf16*)alloc((size_t)ROWS * D_MODEL * 2);
    __bf16* hbf     = (__bf16*)alloc((size_t)ROWS * FFN_H * 2);

    const long nAct = (long)ROWS * D_MODEL;          // 4M
    const dim3 cvtG(4096), cvtB(256);
    const dim3 splitG((unsigned)((nAct + 255) / 256)), splitB(256);
    const dim3 attnG(SEQ / 16, BATCH * N_HEADS), attnB(32);
    const dim3 lnG(ROWS), lnB(256);
    const float scale = 0.125f;                      // 1/sqrt(64)
    auto gemmGrid = [](int M, int N) { return dim3((N + 63) / 64, (M + 63) / 64); };

    // --- convert weights + inputs to bf16 ---
    hipLaunchKernelGGL(f32_to_bf16, cvtG, cvtB, 0, stream, w_qkv, wqkv_bf, (long)D_MODEL * 3 * D_MODEL);
    hipLaunchKernelGGL(f32_to_bf16, cvtG, cvtB, 0, stream, w_so,  wso_bf,  (long)D_MODEL * D_MODEL);
    hipLaunchKernelGGL(f32_to_bf16, cvtG, cvtB, 0, stream, w_kv,  wkv_bf,  (long)D_MODEL * 2 * D_MODEL);
    hipLaunchKernelGGL(f32_to_bf16, cvtG, cvtB, 0, stream, w_q,   wq_bf,   (long)D_MODEL * D_MODEL);
    hipLaunchKernelGGL(f32_to_bf16, cvtG, cvtB, 0, stream, w_co,  wco_bf,  (long)D_MODEL * D_MODEL);
    hipLaunchKernelGGL(f32_to_bf16, cvtG, cvtB, 0, stream, w_f1,  wf1_bf,  (long)D_MODEL * FFN_H);
    hipLaunchKernelGGL(f32_to_bf16, cvtG, cvtB, 0, stream, w_f2,  wf2_bf,  (long)FFN_H * D_MODEL);
    hipLaunchKernelGGL(f32_to_bf16, cvtG, cvtB, 0, stream, x, xbf, nAct);
    hipLaunchKernelGGL(f32_to_bf16, cvtG, cvtB, 0, stream, y, actbf, nAct);   // y_bf

    // --- masked self-attention on y ---
    hipLaunchKernelGGL(gemm_bf16_wmma, gemmGrid(ROWS, 3 * D_MODEL), dim3(128), 0, stream,
                       actbf, wqkv_bf, b_qkv, qkvf, (__bf16*)nullptr, ROWS, 3 * D_MODEL, D_MODEL, 0);
    hipLaunchKernelGGL(split_qkv, splitG, splitB, 0, stream, qkvf, qbf, kbf, vbf);
    hipLaunchKernelGGL(attention_wmma, attnG, attnB, 0, stream, qbf, kbf, vbf, dmask, valsf, scale);
    // faithful no-transpose reshape [B,H,S,hd] -> [B,S,D]: plain convert
    hipLaunchKernelGGL(f32_to_bf16, cvtG, cvtB, 0, stream, valsf, actbf, nAct);
    hipLaunchKernelGGL(gemm_bf16_wmma, gemmGrid(ROWS, D_MODEL), dim3(128), 0, stream,
                       actbf, wso_bf, b_so, zf, (__bf16*)nullptr, ROWS, D_MODEL, D_MODEL, 0);
    hipLaunchKernelGGL(layernorm_kernel, lnG, lnB, 0, stream, zf, y, g1, be1, y1f, y1bf);

    // --- cross attention: kv from x, q from y1 ---
    hipLaunchKernelGGL(gemm_bf16_wmma, gemmGrid(ROWS, 2 * D_MODEL), dim3(128), 0, stream,
                       xbf, wkv_bf, b_kv, qkvf, (__bf16*)nullptr, ROWS, 2 * D_MODEL, D_MODEL, 0);
    hipLaunchKernelGGL(split_kv, splitG, splitB, 0, stream, qkvf, kbf, vbf);
    hipLaunchKernelGGL(gemm_bf16_wmma, gemmGrid(ROWS, D_MODEL), dim3(128), 0, stream,
                       y1bf, wq_bf, b_q, zf, (__bf16*)nullptr, ROWS, D_MODEL, D_MODEL, 0);
    hipLaunchKernelGGL(split_q, splitG, splitB, 0, stream, zf, qbf);
    hipLaunchKernelGGL(attention_wmma, attnG, attnB, 0, stream, qbf, kbf, vbf,
                       (const float*)nullptr, valsf, scale);
    hipLaunchKernelGGL(transpose_vals, splitG, splitB, 0, stream, valsf, actbf);
    hipLaunchKernelGGL(gemm_bf16_wmma, gemmGrid(ROWS, D_MODEL), dim3(128), 0, stream,
                       actbf, wco_bf, b_co, zf, (__bf16*)nullptr, ROWS, D_MODEL, D_MODEL, 0);
    hipLaunchKernelGGL(layernorm_kernel, lnG, lnB, 0, stream, zf, y1f, g2, be2, y2f, y2bf);

    // --- FFN ---
    hipLaunchKernelGGL(gemm_bf16_wmma, gemmGrid(ROWS, FFN_H), dim3(128), 0, stream,
                       y2bf, wf1_bf, b_f1, (float*)nullptr, hbf, ROWS, FFN_H, D_MODEL, 1);
    hipLaunchKernelGGL(gemm_bf16_wmma, gemmGrid(ROWS, D_MODEL), dim3(128), 0, stream,
                       hbf, wf2_bf, b_f2, zf, (__bf16*)nullptr, ROWS, D_MODEL, FFN_H, 0);
    hipLaunchKernelGGL(layernorm_kernel, lnG, lnB, 0, stream, zf, y2f, g3, be3,
                       out, (__bf16*)nullptr);
}
